// GeometryAwarePerceiverAggregator_9740985827768
// MI455X (gfx1250) — matrix operand order repaired and checked by reading, not compiled, not run on hardware
//
#include <hip/hip_runtime.h>
#include <hip/hip_bf16.h>

#define DD 256
#define MM 4096
#define BB 32
#define HH 8
#define NSPLIT 8
#define SPLIT_M 512

// padded LDS strides (bf16 elements): 264*2B = 528B -> 16B aligned, bank stride 4
#define P256 264
#define P1024 1032
#define PP 40

typedef __attribute__((ext_vector_type(16))) __bf16 v16bf_t;
typedef __attribute__((ext_vector_type(8)))  float  v8f_t;

union BFrag { v16bf_t v; unsigned short u[16]; unsigned w[8]; uint4 q[2]; };
union U4W  { uint4 q; unsigned short us[8]; };

struct AttnP { const float *bk,*bo,*bq,*bv,*ln_kv_b,*ln_kv_g,*ln_q_b,*ln_q_g,*wk,*wo,*wq,*wv; };
struct FFP   { const float *b1,*b2,*ln_b,*ln_g,*w1,*w2; };

#define WMMA_BF16(a,b,c) __builtin_amdgcn_wmma_f32_16x16x32_bf16(false,(a),false,(b),(short)0,(c),false,false)

// ---------------- scalar helpers ----------------

__device__ inline unsigned short f2bf(float x){
  union { float f; unsigned u; } v; v.f = x;
  unsigned r = v.u + 0x7FFFu + ((v.u >> 16) & 1u);
  return (unsigned short)(r >> 16);
}
__device__ inline float bf2f(unsigned short x){
  union { unsigned u; float f; } v; v.u = ((unsigned)x) << 16;
  return v.f;
}
__device__ inline float gelu_f(float x){
  return 0.5f * x * (1.0f + erff(x * 0.70710678118654752f));
}

// ---------------- WMMA fragment helpers ----------------

// A 16x32 bf16 tile: lane holds 16 contiguous K (k = 16*hf + s) -> 2x ds_load_b128
__device__ inline v16bf_t load_a_frag(const unsigned short* A, int lda){
  int lane = threadIdx.x & 31;
  int hf = lane >> 4, row = lane & 15;
  BFrag f;
  const unsigned short* p = A + row*lda + 16*hf;
  f.q[0] = *(const uint4*)(p);
  f.q[1] = *(const uint4*)(p + 8);
  return f.v;
}

// B 32x16 bf16 tile from TRANSPOSED weights Wt[n][k] (ldk stride): 16 contiguous per lane
__device__ inline v16bf_t load_b_frag_t(const unsigned short* Wt, int ldk){
  int lane = threadIdx.x & 31;
  int hf = lane >> 4, col = lane & 15;
  BFrag f;
  const unsigned short* p = Wt + col*ldk + 16*hf;
  f.q[0] = *(const uint4*)(p);
  f.q[1] = *(const uint4*)(p + 8);
  return f.v;
}

// C/D 16x16 f32: col = lane&15, row = r + 8*(lane>=16)
__device__ inline v8f_t load_c_frag(const float* C, int ldc){
  int lane = threadIdx.x & 31, hf = lane >> 4, col = lane & 15;
  v8f_t r;
  #pragma unroll
  for (int i = 0; i < 8; ++i) r[i] = C[(i + 8*hf)*ldc + col];
  return r;
}
__device__ inline void store_c_frag(float* C, int ldc, v8f_t v){
  int lane = threadIdx.x & 31, hf = lane >> 4, col = lane & 15;
  #pragma unroll
  for (int i = 0; i < 8; ++i) C[(i + 8*hf)*ldc + col] = v[i];
}

// C(32 x NT) += A(32 x 256, bf16 LDS, padded lda) @ Wt(NT x 256 transposed, ldk)
// blockDim=256 (8 waves); each wave's row-tile is fixed -> A-frags cached in registers.
__device__ inline void block_gemm32(const unsigned short* A, int lda,
                                    const unsigned short* Wt, int ldk,
                                    float* C, int ldc, int NT){
  int wave = threadIdx.x >> 5;
  int tm = (wave & 1) << 4;
  v16bf_t ar[8];
  #pragma unroll
  for (int kk = 0; kk < 8; ++kk) ar[kk] = load_a_frag(A + tm*lda + kk*32, lda);
  int nt = NT >> 4;
  for (int tile = wave; tile < 2*nt; tile += 8){
    int tn = (tile >> 1) << 4;
    v8f_t acc = load_c_frag(C + tm*ldc + tn, ldc);
    const unsigned short* Wb = Wt + tn*ldk;
    #pragma unroll
    for (int kk = 0; kk < 8; ++kk)
      acc = WMMA_BF16(ar[kk], load_b_frag_t(Wb + kk*32, ldk), acc);
    store_c_frag(C + tm*ldc + tn, ldc, acc);
  }
}

// stage (256 x ncols) slice of row-major fp32 weights into TRANSPOSED bf16 LDS [ncols][P256]
__device__ inline void stage_T_cols(const float* src, int srcld, int col0,
                                    int cshift, unsigned short* dst){
  int mask = (1 << cshift) - 1;
  int total = 1 << (cshift + 8);
  for (int i = threadIdx.x; i < total; i += 256){
    int k = i >> cshift, n = i & mask;
    dst[n*P256 + k] = f2bf(src[k*srcld + col0 + n]);
  }
}

// ---------------- LayerNorm rows -> padded bf16 (8 threads/row, width 256) ----------------
// row data held in registers: source (global or LDS) is read exactly once.

__device__ inline void layernorm_rows(const float* __restrict__ src, int srcStride, int nrows,
                                      const float* __restrict__ g, const float* __restrict__ bA,
                                      unsigned short* __restrict__ dst, int dstStride,
                                      float* __restrict__ red){
  __syncthreads();
  int t = threadIdx.x;
  int row = t >> 3, seg = t & 7;
  float xr[32];
  if (row < nrows){
    const float* p = src + row*srcStride + seg*32;
    float s = 0.f, ss = 0.f;
    #pragma unroll
    for (int i = 0; i < 32; ++i) xr[i] = p[i];
    #pragma unroll
    for (int i = 0; i < 32; ++i){ s += xr[i]; ss += xr[i]*xr[i]; }
    red[row*16 + seg] = s;
    red[row*16 + 8 + seg] = ss;
  }
  __syncthreads();
  if (row < nrows){
    float m = 0.f, v = 0.f;
    #pragma unroll
    for (int i = 0; i < 8; ++i){ m += red[row*16 + i]; v += red[row*16 + 8 + i]; }
    m *= (1.f/256.f);
    v = v*(1.f/256.f) - m*m;
    float rstd = rsqrtf(v + 1e-5f);
    #pragma unroll
    for (int i = 0; i < 32; ++i){
      int c = seg*32 + i;
      dst[row*dstStride + c] = f2bf((xr[i]-m)*rstd*g[c] + bA[c]);
    }
  }
  __syncthreads();
}

// ========================= Kernel 1: K/V projection of Z =========================

__global__ void __launch_bounds__(256)
kv_proj_kernel(const float* __restrict__ Z,
               const float* __restrict__ wk, const float* __restrict__ bk,
               const float* __restrict__ wv, const float* __restrict__ bv,
               const float* __restrict__ ln_g, const float* __restrict__ ln_b,
               unsigned short* __restrict__ Kout, unsigned short* __restrict__ Vout){
  extern __shared__ char smem[];
  unsigned short* WkT = (unsigned short*)smem;       // 256*P256
  unsigned short* WvT = WkT + 256*P256;              // 256*P256
  unsigned short* LNT = WvT + 256*P256;              // 32*P256
  float* C   = (float*)(LNT + 32*P256);              // 8192
  float* red = C + 8192;                             // 512
  int t = threadIdx.x;
  int b = blockIdx.x >> 3, ch = blockIdx.x & 7;
  stage_T_cols(wk, 256, 0, 8, WkT);
  stage_T_cols(wv, 256, 0, 8, WvT);
  __syncthreads();
  for (int it = 0; it < 16; ++it){
    size_t rowbase = ((size_t)b*MM + (size_t)ch*SPLIT_M + (size_t)it*32) * DD;
    layernorm_rows(Z + rowbase, 256, 32, ln_g, ln_b, LNT, P256, red);
    for (int i = t; i < 8192; i += 256) C[i] = bk[i & 255];
    __syncthreads();
    block_gemm32(LNT, P256, WkT, P256, C, 256, 256);
    __syncthreads();
    for (int i = t; i < 8192; i += 256) Kout[rowbase + i] = f2bf(C[i]);
    __syncthreads();
    for (int i = t; i < 8192; i += 256) C[i] = bv[i & 255];
    __syncthreads();
    block_gemm32(LNT, P256, WvT, P256, C, 256, 256);
    __syncthreads();
    for (int i = t; i < 8192; i += 256) Vout[rowbase + i] = f2bf(C[i]);
    __syncthreads();
  }
}

// ========================= Kernel 2: Q for cross-attn (batch-shared) =========================

__global__ void __launch_bounds__(256)
qcross_kernel(const float* __restrict__ latents,
              const float* __restrict__ ln_g, const float* __restrict__ ln_b,
              const float* __restrict__ wq, const float* __restrict__ bq,
              unsigned short* __restrict__ Qbf){
  extern __shared__ char smem[];
  unsigned short* LNT = (unsigned short*)smem;   // 32*P256
  unsigned short* Wt  = LNT + 32*P256;           // 256*P256
  float* C   = (float*)(Wt + 256*P256);          // 8192
  float* red = C + 8192;                         // 512
  int t = threadIdx.x;
  layernorm_rows(latents, 256, 32, ln_g, ln_b, LNT, P256, red);
  stage_T_cols(wq, 256, 0, 8, Wt);
  for (int i = t; i < 8192; i += 256) C[i] = bq[i & 255];
  __syncthreads();
  block_gemm32(LNT, P256, Wt, P256, C, 256, 256);
  __syncthreads();
  for (int i = t; i < 8192; i += 256) Qbf[i] = f2bf(C[i] * 0.17677669529663687f); // 1/sqrt(32)
}

// ========================= Kernel 3: flash cross-attention (double-buffered) =========================

// scores B: k-dim = dh contiguous in padded K-tile rows
__device__ inline v16bf_t load_b_frag_kt(const unsigned short* Kt, int h, int mbase){
  int lane = threadIdx.x & 31, hf = lane >> 4, col = lane & 15;
  BFrag f;
  const unsigned short* p = Kt + (mbase + col)*P256 + h*32 + 16*hf;
  f.q[0] = *(const uint4*)(p);
  f.q[1] = *(const uint4*)(p + 8);
  return f.v;
}
// P@V B from transposed V-tile Vt[d][m] (stride PP): contiguous in m
__device__ inline v16bf_t load_b_frag_vt(const unsigned short* Vt, int dbase){
  int lane = threadIdx.x & 31, hf = lane >> 4, col = lane & 15;
  BFrag f;
  const unsigned short* p = Vt + (dbase + col)*PP + 16*hf;
  f.q[0] = *(const uint4*)(p);
  f.q[1] = *(const uint4*)(p + 8);
  return f.v;
}

__device__ inline void bias_softmax_update(
    v8f_t& s0, v8f_t& s1, v8f_t& o0, v8f_t& o1,
    float* mr, float* lr,
    const float* anc, int latbase, float cb,
    float qx0, float qy0, float qz0,
    float qx1, float qy1, float qz1){
  #pragma unroll
  for (int r = 0; r < 8; ++r){
    int lat = latbase + r;
    float ax = anc[lat*3+0], ay = anc[lat*3+1], az = anc[lat*3+2];
    float dx = ax-qx0, dy = ay-qy0, dz = az-qz0;
    float v0 = s0[r] + cb*(dx*dx + dy*dy + dz*dz);
    dx = ax-qx1; dy = ay-qy1; dz = az-qz1;
    float v1 = s1[r] + cb*(dx*dx + dy*dy + dz*dz);
    float mx = fmaxf(v0, v1);
    #pragma unroll
    for (int off = 1; off < 16; off <<= 1) mx = fmaxf(mx, __shfl_xor(mx, off));
    float mnew = fmaxf(mr[r], mx);
    float sc = __expf(mr[r] - mnew);
    float p0 = __expf(v0 - mnew);
    float p1 = __expf(v1 - mnew);
    float rs = p0 + p1;
    #pragma unroll
    for (int off = 1; off < 16; off <<= 1) rs += __shfl_xor(rs, off);
    lr[r] = lr[r]*sc + rs;
    mr[r] = mnew;
    s0[r] = p0; s1[r] = p1;
    o0[r] *= sc; o1[r] *= sc;
  }
}

__global__ void __launch_bounds__(256)
flash_cross_kernel(const unsigned short* __restrict__ Qbf,
                   const unsigned short* __restrict__ Kbf,
                   const unsigned short* __restrict__ Vbf,
                   const float* __restrict__ q_coords,
                   const float* __restrict__ anchors,
                   const float* __restrict__ beta_p,
                   const float* __restrict__ sigma_p,
                   float* __restrict__ pO, float* __restrict__ pm, float* __restrict__ pl){
  extern __shared__ char smem[];
  unsigned short* Qs  = (unsigned short*)smem;  // 32*P256
  unsigned short* Kt0 = Qs + 32*P256;           // 32*P256
  unsigned short* Kt1 = Kt0 + 32*P256;          // 32*P256
  unsigned short* Vt0 = Kt1 + 32*P256;          // 256*PP
  unsigned short* Vt1 = Vt0 + 256*PP;           // 256*PP
  unsigned short* Pst = Vt1 + 256*PP;           // 8 waves * 32*PP
  float* qc0 = (float*)(Pst + 8*32*PP);         // 96
  float* qc1 = qc0 + 96;                        // 96
  float* anc = qc1 + 96;                        // 96

  const int t = threadIdx.x;
  const int lane = t & 31, wave = t >> 5;
  const int col = lane & 15, hf = lane >> 4;
  const int b = blockIdx.x >> 3, sp = blockIdx.x & 7;
  const int h = wave;

  const size_t kvbase = ((size_t)b*MM + (size_t)sp*SPLIT_M) * DD;
  const size_t qcbase = ((size_t)b*MM + (size_t)sp*SPLIT_M) * 3;

  unsigned kreg[16];
  uint4 vreg[4];
  float qcreg = 0.f;

  auto issue_loads = [&](int m0){
    const unsigned* Ksrc = (const unsigned*)(Kbf + kvbase + (size_t)m0*DD);
    const uint4*   Vsrc = (const uint4*)(Vbf + kvbase + (size_t)m0*DD);
    #pragma unroll
    for (int j = 0; j < 16; ++j) kreg[j] = Ksrc[t + j*256];
    #pragma unroll
    for (int j = 0; j < 4; ++j) vreg[j] = Vsrc[t + j*256];
    if (t < 96) qcreg = q_coords[qcbase + (size_t)m0*3 + t];
  };
  auto store_tile = [&](unsigned short* Ktb, unsigned short* Vtb, float* qcb){
    #pragma unroll
    for (int j = 0; j < 16; ++j){
      int i = t + j*256;
      ((unsigned*)(Ktb + (i >> 7)*P256))[i & 127] = kreg[j];
    }
    #pragma unroll
    for (int j = 0; j < 4; ++j){
      int i = t + j*256;
      U4W u; u.q = vreg[j];
      int m = i >> 5, d0 = (i & 31)*8;
      #pragma unroll
      for (int jj = 0; jj < 8; ++jj) Vtb[(d0 + jj)*PP + m] = u.us[jj];
    }
    if (t < 96) qcb[t] = qcreg;
  };

  // prologue: Q + anchors + first tile
  for (int i = t; i < 4096; i += 256){
    int row = i >> 7, c2 = i & 127;
    ((unsigned*)(Qs + row*P256))[c2] = ((const unsigned*)Qbf)[i];
  }
  if (t < 96) anc[t] = anchors[t];
  float sg = log1pf(__expf(sigma_p[0]));         // softplus
  float cb = -beta_p[0] / (2.f*sg*sg);
  issue_loads(0);
  store_tile(Kt0, Vt0, qc0);
  __syncthreads();

  v16bf_t a0 = load_a_frag(Qs + h*32, P256);
  v16bf_t a1 = load_a_frag(Qs + 16*P256 + h*32, P256);

  v8f_t o00 = {}, o01 = {}, o10 = {}, o11 = {};
  float mr0[8], mr1[8], lr0[8], lr1[8];
  #pragma unroll
  for (int r = 0; r < 8; ++r){ mr0[r] = -3.0e38f; mr1[r] = -3.0e38f; lr0[r] = 0.f; lr1[r] = 0.f; }

  for (int it = 0; it < 16; ++it){
    unsigned short* Ktc = (it & 1) ? Kt1 : Kt0;
    unsigned short* Vtc = (it & 1) ? Vt1 : Vt0;
    float* qcc = (it & 1) ? qc1 : qc0;
    if (it < 15) issue_loads((it + 1) * 32);   // overlap HBM with compute

    float qx0 = qcc[col*3+0], qy0 = qcc[col*3+1], qz0 = qcc[col*3+2];
    float qx1 = qcc[(16+col)*3+0], qy1 = qcc[(16+col)*3+1], qz1 = qcc[(16+col)*3+2];

    v16bf_t bk0 = load_b_frag_kt(Ktc, h, 0);
    v16bf_t bk1 = load_b_frag_kt(Ktc, h, 16);
    v8f_t z = {};
    v8f_t s00 = WMMA_BF16(a0, bk0, z);
    v8f_t s01 = WMMA_BF16(a0, bk1, z);
    v8f_t s10 = WMMA_BF16(a1, bk0, z);
    v8f_t s11 = WMMA_BF16(a1, bk1, z);

    bias_softmax_update(s00, s01, o00, o01, mr0, lr0, anc,      8*hf, cb, qx0,qy0,qz0, qx1,qy1,qz1);
    bias_softmax_update(s10, s11, o10, o11, mr1, lr1, anc, 16 + 8*hf, cb, qx0,qy0,qz0, qx1,qy1,qz1);

    unsigned short* Pw = Pst + wave*(32*PP);
    #pragma unroll
    for (int r = 0; r < 8; ++r){
      int la = r + 8*hf, lb = 16 + r + 8*hf;
      Pw[la*PP + col]      = f2bf(s00[r]);
      Pw[la*PP + 16 + col] = f2bf(s01[r]);
      Pw[lb*PP + col]      = f2bf(s10[r]);
      Pw[lb*PP + 16 + col] = f2bf(s11[r]);
    }
    __syncthreads();
    if (it < 15) store_tile((it & 1) ? Kt0 : Kt1, (it & 1) ? Vt0 : Vt1, (it & 1) ? qc0 : qc1);
    v16bf_t pa0 = load_a_frag(Pw, PP);
    v16bf_t pa1 = load_a_frag(Pw + 16*PP, PP);
    v16bf_t bv0 = load_b_frag_vt(Vtc, h*32 + 0);
    v16bf_t bv1 = load_b_frag_vt(Vtc, h*32 + 16);
    o00 = WMMA_BF16(pa0, bv0, o00);
    o01 = WMMA_BF16(pa0, bv1, o01);
    o10 = WMMA_BF16(pa1, bv0, o10);
    o11 = WMMA_BF16(pa1, bv1, o11);
    __syncthreads();
  }

  const size_t pbase = (((size_t)b*NSPLIT + sp)*HH + h) * 1024;
  #pragma unroll
  for (int r = 0; r < 8; ++r){
    int la = r + 8*hf, lb = 16 + r + 8*hf;
    pO[pbase + la*32 + col]      = o00[r];
    pO[pbase + la*32 + 16 + col] = o01[r];
    pO[pbase + lb*32 + col]      = o10[r];
    pO[pbase + lb*32 + 16 + col] = o11[r];
  }
  if (col == 0){
    const size_t mb = (((size_t)b*NSPLIT + sp)*HH + h) * 32;
    #pragma unroll
    for (int r = 0; r < 8; ++r){
      pm[mb + r + 8*hf] = mr0[r];       pl[mb + r + 8*hf] = lr0[r];
      pm[mb + 16 + r + 8*hf] = mr1[r];  pl[mb + 16 + r + 8*hf] = lr1[r];
    }
  }
}

// ========================= Kernel 4: LSE combine + cross output projection =========================

__global__ void __launch_bounds__(256)
combine_kernel(const float* __restrict__ pO, const float* __restrict__ pm, const float* __restrict__ pl,
               const float* __restrict__ latents, const float* __restrict__ wo, const float* __restrict__ bo,
               float* __restrict__ lat1){
  extern __shared__ char smem[];
  float* Of = (float*)smem;                               // 32*P256 f32
  unsigned short* Ob = (unsigned short*)(Of + 32*P256);   // 32*P256 u16
  unsigned short* Wt = Ob + 32*P256;                      // 256*P256
  float* C = (float*)(Wt + 256*P256);                     // 8192
  int t = threadIdx.x, b = blockIdx.x;
  int h = t >> 5, lat = t & 31;
  float mmax = -3.0e38f;
  #pragma unroll
  for (int s = 0; s < NSPLIT; ++s)
    mmax = fmaxf(mmax, pm[(((size_t)b*NSPLIT + s)*HH + h)*32 + lat]);
  float wgt[NSPLIT]; float L = 0.f;
  #pragma unroll
  for (int s = 0; s < NSPLIT; ++s){
    size_t idx = (((size_t)b*NSPLIT + s)*HH + h)*32 + lat;
    wgt[s] = __expf(pm[idx] - mmax);
    L += pl[idx]*wgt[s];
  }
  float invL = 1.f / L;
  for (int d = 0; d < 32; ++d){
    float acc = 0.f;
    #pragma unroll
    for (int s = 0; s < NSPLIT; ++s)
      acc += pO[(((size_t)b*NSPLIT + s)*HH + h)*1024 + lat*32 + d] * wgt[s];
    Of[lat*P256 + h*32 + d] = acc * invL;
  }
  __syncthreads();
  for (int i = t; i < 8192; i += 256){
    int row = i >> 8, c = i & 255;
    Ob[row*P256 + c] = f2bf(Of[row*P256 + c]);
  }
  stage_T_cols(wo, 256, 0, 8, Wt);
  for (int i = t; i < 8192; i += 256) C[i] = latents[i] + bo[i & 255];
  __syncthreads();
  block_gemm32(Ob, P256, Wt, P256, C, 256, 256);
  __syncthreads();
  for (int i = t; i < 8192; i += 256) lat1[(size_t)b*8192 + i] = C[i];
}

// ========================= Kernel 5: latent stack tail =========================

// C = A(32x256 padded bf16) @ wg(256x256) + bg ; weights staged in two 128-col chunks
__device__ inline void gemm_into(const unsigned short* A, const float* wg, const float* bg,
                                 float* C, unsigned short* Wt){
  int t = threadIdx.x;
  __syncthreads();
  for (int i = t; i < 8192; i += 256) C[i] = bg[i & 255];
  for (int nh = 0; nh < 2; ++nh){
    __syncthreads();
    stage_T_cols(wg, 256, nh*128, 7, Wt);
    __syncthreads();
    block_gemm32(A, P256, Wt, P256, C + nh*128, 256, 128);
  }
  __syncthreads();
}

__device__ void ff_block(float* x, float* C, unsigned short* A0, unsigned short* Hb,
                         unsigned short* Wt, float* red, const FFP& p){
  int t = threadIdx.x;
  layernorm_rows(x, 256, 32, p.ln_g, p.ln_b, A0, P256, red);
  for (int nc = 0; nc < 8; ++nc){
    __syncthreads();
    stage_T_cols(p.w1, 1024, nc*128, 7, Wt);
    for (int i = t; i < 4096; i += 256) C[i] = p.b1[nc*128 + (i & 127)];
    __syncthreads();
    block_gemm32(A0, P256, Wt, P256, C, 128, 128);
    __syncthreads();
    for (int i = t; i < 4096; i += 256){
      int row = i >> 7, cc = i & 127;
      Hb[row*P1024 + nc*128 + cc] = f2bf(gelu_f(C[i]));
    }
  }
  __syncthreads();
  for (int i = t; i < 8192; i += 256) C[i] = p.b2[i & 255];
  for (int kc = 0; kc < 4; ++kc){
    for (int nh = 0; nh < 2; ++nh){
      __syncthreads();
      stage_T_cols(p.w2 + (size_t)kc*256*256, 256, nh*128, 7, Wt);
      __syncthreads();
      block_gemm32(Hb + kc*256, P1024, Wt, P256, C + nh*128, 256, 128);
    }
  }
  __syncthreads();
  for (int i = t; i < 8192; i += 256) x[i] += C[i];
  __syncthreads();
}

__device__ void self_attn_block(float* x, float* C, unsigned short* A0, unsigned short* A1,
                                unsigned short* Hb, unsigned short* Wt, float* red, const AttnP& p){
  int t = threadIdx.x;
  layernorm_rows(x, 256, 32, p.ln_q_g,  p.ln_q_b,  A0, P256, red);
  layernorm_rows(x, 256, 32, p.ln_kv_g, p.ln_kv_b, A1, P256, red);
  unsigned short* Qb = Hb;                 // each [32][P256]
  unsigned short* Kb = Hb + 32*P256;
  unsigned short* Vb = Hb + 64*P256;
  gemm_into(A0, p.wq, p.bq, C, Wt);
  for (int i = t; i < 8192; i += 256) Qb[(i>>8)*P256 + (i&255)] = f2bf(C[i] * 0.17677669529663687f);
  gemm_into(A1, p.wk, p.bk, C, Wt);
  for (int i = t; i < 8192; i += 256) Kb[(i>>8)*P256 + (i&255)] = f2bf(C[i]);
  gemm_into(A1, p.wv, p.bv, C, Wt);
  for (int i = t; i < 8192; i += 256) Vb[(i>>8)*P256 + (i&255)] = f2bf(C[i]);
  __syncthreads();
  int h = t >> 5, q = t & 31;
  float qr[32];
  #pragma unroll
  for (int d = 0; d < 32; ++d) qr[d] = bf2f(Qb[q*P256 + h*32 + d]);
  float sc_[32];
  #pragma unroll
  for (int m = 0; m < 32; ++m){
    float acc = 0.f;
    #pragma unroll
    for (int d = 0; d < 32; ++d) acc += qr[d]*bf2f(Kb[m*P256 + h*32 + d]);
    sc_[m] = acc;
  }
  float mx = -3.0e38f;
  #pragma unroll
  for (int m = 0; m < 32; ++m) mx = fmaxf(mx, sc_[m]);
  float L = 0.f;
  #pragma unroll
  for (int m = 0; m < 32; ++m){ sc_[m] = __expf(sc_[m] - mx); L += sc_[m]; }
  float invL = 1.f / L;
  #pragma unroll
  for (int d = 0; d < 32; ++d){
    float acc = 0.f;
    #pragma unroll
    for (int m = 0; m < 32; ++m) acc += sc_[m]*bf2f(Vb[m*P256 + h*32 + d]);
    A0[q*P256 + h*32 + d] = f2bf(acc*invL);
  }
  __syncthreads();
  gemm_into(A0, p.wo, p.bo, C, Wt);
  for (int i = t; i < 8192; i += 256) x[i] += C[i];
  __syncthreads();
}

__device__ void pool_and_proj(float* x, float* C, unsigned short* A1, unsigned short* Hb,
                              unsigned short* Wt, float* red,
                              const AttnP& p, const float* qglob,
                              const float* png, const float* pnb,
                              const float* pw1, const float* pb1,
                              const float* pw2, const float* pb2,
                              int b, float* out){
  int t = threadIdx.x;
  layernorm_rows(x, 256, 32, p.ln_kv_g, p.ln_kv_b, A1, P256, red);
  unsigned short* Kb = Hb;
  unsigned short* Vb = Hb + 32*P256;
  gemm_into(A1, p.wk, p.bk, C, Wt);
  for (int i = t; i < 8192; i += 256) Kb[(i>>8)*P256 + (i&255)] = f2bf(C[i]);
  gemm_into(A1, p.wv, p.bv, C, Wt);
  for (int i = t; i < 8192; i += 256) Vb[(i>>8)*P256 + (i&255)] = f2bf(C[i]);
  __syncthreads();
  // LN of the single q_glob row
  float xv = qglob[t];
  red[t] = xv;
  __syncthreads();
  if (t == 0){
    float s = 0.f, ss = 0.f;
    for (int k = 0; k < 256; ++k){ float u = red[k]; s += u; ss += u*u; }
    float m = s*(1.f/256.f);
    red[256] = m; red[257] = ss*(1.f/256.f) - m*m;
  }
  __syncthreads();
  float qn = (xv - red[256])*rsqrtf(red[257] + 1e-5f)*p.ln_q_g[t] + p.ln_q_b[t];
  __syncthreads();
  red[t] = qn;
  __syncthreads();
  float q1 = p.bq[t];
  for (int k = 0; k < 256; ++k) q1 += red[k]*p.wq[k*256 + t];
  q1 *= 0.17677669529663687f;
  red[256 + t] = q1;
  __syncthreads();
  // wave h, lane = key index m
  int h = t >> 5, lane = t & 31;
  float s = 0.f;
  for (int d = 0; d < 32; ++d) s += red[256 + h*32 + d]*bf2f(Kb[lane*P256 + h*32 + d]);
  float mx = s;
  #pragma unroll
  for (int off = 1; off < 32; off <<= 1) mx = fmaxf(mx, __shfl_xor(mx, off));
  float e = __expf(s - mx);
  float L = e;
  #pragma unroll
  for (int off = 1; off < 32; off <<= 1) L += __shfl_xor(L, off);
  red[512 + h*32 + lane] = e / L;
  __syncthreads();
  // lane = d index now
  float acc = 0.f;
  for (int m = 0; m < 32; ++m) acc += red[512 + h*32 + m]*bf2f(Vb[m*P256 + h*32 + lane]);
  red[768 + h*32 + lane] = acc;
  __syncthreads();
  float z = p.bo[t] + qglob[t];
  for (int k = 0; k < 256; ++k) z += red[768 + k]*p.wo[k*256 + t];
  __syncthreads();
  red[t] = z;
  __syncthreads();
  if (t == 0){
    float sA = 0.f, sS = 0.f;
    for (int k = 0; k < 256; ++k){ float u = red[k]; sA += u; sS += u*u; }
    float m = sA*(1.f/256.f);
    red[256] = m; red[257] = sS*(1.f/256.f) - m*m;
  }
  __syncthreads();
  float zf = (z - red[256])*rsqrtf(red[257] + 1e-5f)*png[t] + pnb[t];
  out[(size_t)b*256 + t] = zf;
  red[512 + t] = zf;
  __syncthreads();
  float h1 = pb1[t];
  for (int k = 0; k < 256; ++k) h1 += red[512 + k]*pw1[k*256 + t];
  h1 = gelu_f(h1);
  red[768 + t] = h1;
  __syncthreads();
  if (t < 128){
    float zp = pb2[t];
    for (int k = 0; k < 256; ++k) zp += red[768 + k]*pw2[k*128 + t];
    out[(size_t)BB*256 + (size_t)b*128 + t] = zp;
  }
}

__global__ void __launch_bounds__(256)
tail_kernel(const float* __restrict__ lat1,
            FFP cff, AttnP s0a, FFP s0f, AttnP s1a, FFP s1f,
            AttnP pool, const float* __restrict__ qglob,
            const float* __restrict__ png, const float* __restrict__ pnb,
            const float* __restrict__ pw1, const float* __restrict__ pb1,
            const float* __restrict__ pw2, const float* __restrict__ pb2,
            float* __restrict__ out){
  extern __shared__ char smem[];
  float* x = (float*)smem;                              // 8192 f32
  float* C = x + 8192;                                  // 8192 f32
  unsigned short* A0 = (unsigned short*)(C + 8192);     // 32*P256
  unsigned short* A1 = A0 + 32*P256;                    // 32*P256
  unsigned short* Hb = A1 + 32*P256;                    // 32*P1024
  unsigned short* Wt = Hb + 32*P1024;                   // 128*P256
  float* red = (float*)(Wt + 128*P256);                 // 1024 f32
  int t = threadIdx.x;
  int b = blockIdx.x;
  for (int i = t; i < 8192; i += 256) x[i] = lat1[(size_t)b*8192 + i];
  __syncthreads();
  ff_block(x, C, A0, Hb, Wt, red, cff);
  self_attn_block(x, C, A0, A1, Hb, Wt, red, s0a);
  ff_block(x, C, A0, Hb, Wt, red, s0f);
  self_attn_block(x, C, A0, A1, Hb, Wt, red, s1a);
  ff_block(x, C, A0, Hb, Wt, red, s1f);
  pool_and_proj(x, C, A1, Hb, Wt, red, pool, qglob, png, pnb, pw1, pb1, pw2, pb2, b, out);
}

// ========================= host =========================

static inline AttnP mkAttn(void* const* d, int i0){
  AttnP a;
  a.bk = (const float*)d[i0+0];  a.bo = (const float*)d[i0+1];
  a.bq = (const float*)d[i0+2];  a.bv = (const float*)d[i0+3];
  a.ln_kv_b = (const float*)d[i0+4]; a.ln_kv_g = (const float*)d[i0+5];
  a.ln_q_b  = (const float*)d[i0+6]; a.ln_q_g  = (const float*)d[i0+7];
  a.wk = (const float*)d[i0+8];  a.wo = (const float*)d[i0+9];
  a.wq = (const float*)d[i0+10]; a.wv = (const float*)d[i0+11];
  return a;
}
static inline FFP mkFF(void* const* d, int i0){
  FFP f;
  f.b1 = (const float*)d[i0+0]; f.b2 = (const float*)d[i0+1];
  f.ln_b = (const float*)d[i0+2]; f.ln_g = (const float*)d[i0+3];
  f.w1 = (const float*)d[i0+4]; f.w2 = (const float*)d[i0+5];
  return f;
}

extern "C" void kernel_launch(void* const* d_in, const int* in_sizes, int n_in,
                              void* d_out, int out_size, void* d_ws, size_t ws_size,
                              hipStream_t stream){
  (void)in_sizes; (void)n_in; (void)out_size; (void)ws_size;
  // jax pytree (sorted-key) flat order:
  const float* Z       = (const float*)d_in[0];
  const float* qcoords = (const float*)d_in[1];
  const float* anchors = (const float*)d_in[2];
  const float* beta    = (const float*)d_in[3];
  AttnP cross = mkAttn(d_in, 4);
  FFP   cff   = mkFF(d_in, 16);
  AttnP s0a   = mkAttn(d_in, 22);
  FFP   s0f   = mkFF(d_in, 34);
  AttnP s1a   = mkAttn(d_in, 40);
  FFP   s1f   = mkFF(d_in, 52);
  const float* latents = (const float*)d_in[58];
  AttnP pool  = mkAttn(d_in, 59);
  const float* pn_b  = (const float*)d_in[71];
  const float* pn_g  = (const float*)d_in[72];
  const float* pj_b1 = (const float*)d_in[73];
  const float* pj_b2 = (const float*)d_in[74];
  const float* pj_w1 = (const float*)d_in[75];
  const float* pj_w2 = (const float*)d_in[76];
  const float* qglob = (const float*)d_in[77];
  const float* sigma = (const float*)d_in[78];
  float* out = (float*)d_out;

  // workspace layout
  unsigned short* Kbf = (unsigned short*)d_ws;
  unsigned short* Vbf = Kbf + (size_t)BB*MM*DD;
  unsigned short* Qbf = Vbf + (size_t)BB*MM*DD;
  float* pO   = (float*)(Qbf + 8192);
  float* pmv  = pO  + (size_t)BB*NSPLIT*HH*1024;
  float* plv  = pmv + (size_t)BB*NSPLIT*HH*32;
  float* lat1 = plv + (size_t)BB*NSPLIT*HH*32;

  const size_t SM_KV = (2*256*P256 + 32*P256)*2 + 32768 + 2048;             // 322048
  const size_t SM_QC = (32*P256 + 256*P256)*2 + 32768 + 2048;               // 186880
  const size_t SM_FL = (3*32*P256 + 2*256*PP + 8*32*PP)*2 + 3*96*4;         // 113280
  const size_t SM_CB = 32*P256*4 + (32*P256 + 256*P256)*2 + 32768;          // 218624
  const size_t SM_TL = 32768 + 32768 + (32*P256*2 + 32*P1024 + 128*P256)*2 + 4096; // 237056

  kv_proj_kernel<<<dim3(BB*NSPLIT), dim3(256), SM_KV, stream>>>(
      Z, cross.wk, cross.bk, cross.wv, cross.bv, cross.ln_kv_g, cross.ln_kv_b, Kbf, Vbf);
  qcross_kernel<<<dim3(1), dim3(256), SM_QC, stream>>>(
      latents, cross.ln_q_g, cross.ln_q_b, cross.wq, cross.bq, Qbf);
  flash_cross_kernel<<<dim3(BB*NSPLIT), dim3(256), SM_FL, stream>>>(
      Qbf, Kbf, Vbf, qcoords, anchors, beta, sigma, pO, pmv, plv);
  combine_kernel<<<dim3(BB), dim3(256), SM_CB, stream>>>(
      pO, pmv, plv, latents, cross.wo, cross.bo, lat1);
  tail_kernel<<<dim3(BB), dim3(256), SM_TL, stream>>>(
      lat1, cff, s0a, s0f, s1a, s1f, pool, qglob, pn_g, pn_b, pj_w1, pj_b1, pj_w2, pj_b2, out);
}
